// DLSTMLayer_28793460752972
// MI455X (gfx1250) — matrix operand order repaired and checked by reading, not compiled, not run on hardware
//
#include <hip/hip_runtime.h>

// ---------------------------------------------------------------------------
// dLSTM layer for MI455X (gfx1250): bf16 WMMA recurrence, fused input proj,
// async global->LDS staging of the recurrent weight tiles (ASYNCcnt path).
//   x:[128,512,64] W:[64,2048] U:[512,2048] b:[2048] Wd:[64,512]
//   out = hidden_seq [128,512,512] ++ h_T [128,512] ++ c_T [128,512]
// ---------------------------------------------------------------------------

typedef __attribute__((ext_vector_type(16))) __bf16       v16bf;
typedef __attribute__((ext_vector_type(8)))  float        v8f;
typedef __attribute__((ext_vector_type(4)))  unsigned int u32x4;

#define B_   128
#define T_   512
#define E_   64
#define H_   512
#define FH_  2048
#define KT_G 18    // (512+64)/32 k-tiles of combined [U;W]
#define NT_G 128   // 2048/16 n-tiles of combined [U;W]

// Load 16 contiguous bf16 (32 bytes, 16B-aligned) as a WMMA operand vector.
// Works for both global and LDS (addrspace inferred -> ds_load_b128 x2).
__device__ __forceinline__ v16bf load16bf(const __bf16* p) {
  union { u32x4 u[2]; v16bf v; } tmp;
  tmp.u[0] = *(const u32x4*)(p);
  tmp.u[1] = *(const u32x4*)(p + 8);
  return tmp.v;
}

__device__ __forceinline__ float sigm(float x) {
  return 1.0f / (1.0f + __expf(-x));
}
__device__ __forceinline__ float tanh_fast(float x) {
  float e = __expf(-2.0f * x);
  return (1.0f - e) / (1.0f + e);
}

// CDNA5 async copy: global -> LDS, 16 bytes per lane, tracked by ASYNCcnt.
__device__ __forceinline__ void async_ld_b128(unsigned lds_off,
                                              const void* gptr) {
  asm volatile("global_load_async_to_lds_b128 %0, %1, off"
               :: "v"(lds_off), "v"((unsigned long long)(uintptr_t)gptr)
               : "memory");
}
__device__ __forceinline__ void wait_async0() {
  asm volatile("s_wait_asynccnt 0x0" ::: "memory");
}
// Drain our DS reads. Needed before the next async DMA may overwrite a stage
// buffer: async-to-LDS writes are NOT ordered against in-flight ds_loads
// (ISA §4.1), so every wave drains its reads before signaling the barrier
// that releases the buffer for the next DMA.
__device__ __forceinline__ void wait_ds0() {
  asm volatile("s_wait_dscnt 0x0" ::: "memory");
}
// Workgroup barrier WITHOUT the implicit s_wait_loadcnt/dscnt 0 drain that
// __syncthreads() emits: ordering of the LDS stage handoff is done explicitly
// with s_wait_asynccnt / s_wait_dscnt, so the A-operand global prefetch can
// stay in flight across the barrier.
__device__ __forceinline__ void wg_barrier() {
  asm volatile("s_barrier_signal -1\n\ts_barrier_wait -1" ::: "memory");
}

// --------------------------- prep kernels ----------------------------------

// Swizzle G = [U;W] (576 x 2048 fp32) into bf16 WMMA-B tiles (32K x 16N).
// Tile (ntg, kt) at element offset (ntg*KT_G + kt)*512; within a tile, lane l
// holds 16 contiguous bf16: N = ntg*16 + (l&15), K = kt*32 + (l<16?0:16) + q.
__global__ void swizzle_G(const float* __restrict__ U, const float* __restrict__ W,
                          void* __restrict__ outp) {
  __bf16* out = (__bf16*)outp;
  int e = blockIdx.x * blockDim.x + threadIdx.x;
  if (e >= NT_G * KT_G * 512) return;
  int tile = e >> 9, rem = e & 511;
  int lane = rem >> 4, q = rem & 15;
  int kt = tile % KT_G, ntg = tile / KT_G;
  int koff = (lane < 16) ? 0 : 16;
  int nn = ntg * 16 + (lane & 15);
  int k = kt * 32 + koff + q;
  float s = (k < H_) ? U[(size_t)k * FH_ + nn] : W[(size_t)(k - H_) * FH_ + nn];
  out[e] = (__bf16)s;
}

// Swizzle Wd (64 x 512 fp32) into bf16 WMMA-B tiles; tile (ntw, kt), kt in 0..1.
__global__ void swizzle_Wd(const float* __restrict__ Wd, void* __restrict__ outp) {
  __bf16* out = (__bf16*)outp;
  int e = blockIdx.x * blockDim.x + threadIdx.x;
  if (e >= 32 * 2 * 512) return;
  int tile = e >> 9, rem = e & 511;
  int lane = rem >> 4, q = rem & 15;
  int kt = tile & 1, ntw = tile >> 1;
  int koff = (lane < 16) ? 0 : 16;
  int nn = ntw * 16 + (lane & 15);
  int k = kt * 32 + koff + q;
  out[e] = (__bf16)Wd[(size_t)k * H_ + nn];
}

// x -> bf16, and xd = x[t]-x[t-1] (0 at t==0) -> bf16, row-major [b,t,e].
__global__ void conv_x(const float* __restrict__ x, void* __restrict__ xbp,
                       void* __restrict__ xdbp) {
  __bf16* xb  = (__bf16*)xbp;
  __bf16* xdb = (__bf16*)xdbp;
  int e = blockIdx.x * blockDim.x + threadIdx.x;
  if (e >= B_ * T_ * E_) return;
  int t = (e >> 6) & (T_ - 1);
  float xv = x[e];
  xb[e]  = (__bf16)xv;
  xdb[e] = (t == 0) ? (__bf16)0.0f : (__bf16)(xv - x[e - E_]);
}

__global__ void zero_u32(unsigned int* __restrict__ p, int n) {
  int i = blockIdx.x * blockDim.x + threadIdx.x;
  if (i < n) p[i] = 0u;
}

// --------------------------- recurrent step --------------------------------
// grid = 32 (one WG per 16-col tile of H), block = 256 (8 waves = 8 M-tiles).
// Per k-tile: WG async-DMAs the next 4 gate B-tiles (4KB) into LDS stage
// buffer while waves run 4 WMMAs each against the current stage from LDS.
// Handoff protocol per k-tile:
//   s_wait_asynccnt 0  -> my DMA lanes for stage cur landed
//   barrier            -> everyone's landed; everyone drained reads of 1-cur
//   8x ds_load_b128    -> stage cur B fragments into registers
//   s_wait_dscnt 0     -> my reads drained (buffer reusable after my next signal)
//   async DMA 1-cur    -> overlaps with...
//   4x WMMA            -> ...the current k-tile's math
__global__ __launch_bounds__(256) void dlstm_step(
    const void* __restrict__ Gp, const void* __restrict__ Wdp,
    const void* __restrict__ xp, const void* __restrict__ xdp,
    const float* __restrict__ bias,
    const void* __restrict__ hinp, void* __restrict__ houtp,
    float* __restrict__ c_buf, float* __restrict__ out_h, int t) {
  const __bf16* G_sw  = (const __bf16*)Gp;
  const __bf16* Wd_sw = (const __bf16*)Wdp;
  const __bf16* x_bf  = (const __bf16*)xp;
  const __bf16* xd_bf = (const __bf16*)xdp;
  const __bf16* h_in  = (const __bf16*)hinp;
  __bf16*       h_out = (__bf16*)houtp;

  // Double-buffered B stage: 2 stages x 4 gate tiles x 1KB.
  __shared__ __align__(16) unsigned char lds_b[2 * 4 * 1024];

  const int tid    = threadIdx.x;
  const int lane   = tid & 31;
  const int wave   = tid >> 5;     // M-tile 0..7
  const int nt     = blockIdx.x;   // H column tile 0..31
  const int m_base = wave * 16;

  const unsigned lds_base = (unsigned)(uintptr_t)(&lds_b[0]);

  // Cooperative async issue of one k-tile stage (4 tiles, 16B per thread).
  const int  bg   = tid >> 6;          // gate 0..3 (64 threads per tile)
  const int  boff = (tid & 63) * 16;   // byte offset within 1KB tile
  const char* Gbytes = (const char*)G_sw;
  auto issueB = [&](int kt, int stage) {
    const char* gp =
        Gbytes + (((size_t)(bg * 32 + nt) * KT_G + kt) << 10) + boff;
    async_ld_b128(lds_base + stage * 4096 + bg * 1024 + boff, gp);
  };

  v8f acc[4] = {};  // i, f, g, o accumulators (f32)

  // A-operand k order used here (contiguous, same as B swizzle): lanes 0-15 -> K 0-15,
  // lanes 16-31 -> K 16-31, row M = lane&15.
  const int m_row  = m_base + (lane & 15);
  const int a_koff = (lane < 16) ? 0 : 16;

  const __bf16* hrow = h_in + (size_t)m_row * H_ + a_koff;
  const __bf16* xrow = x_bf + ((size_t)m_row * T_ + t) * E_ + a_koff;

  auto loadA = [&](int kt) -> v16bf {
    const __bf16* p = (kt < 16) ? (hrow + kt * 32) : (xrow + (kt - 16) * 32);
    return load16bf(p);
  };

  // Prologue: stage 0 in flight, first A fragment in registers.
  issueB(0, 0);
  v16bf a_cur = loadA(0);

  // --- [h_{t-1} | x_t] @ [U; W] : 18 k-tiles, LDS double-buffered B ---
#pragma unroll
  for (int kt = 0; kt < KT_G; ++kt) {
    const int cur = kt & 1;
    wait_async0();      // my stage-kt DMA lanes landed in LDS
    wg_barrier();       // all landed; all reads of buffer 1-cur drained
    // Pull all 4 gate B fragments into distinct registers (8 ds_load_b128).
    v16bf bv[4];
#pragma unroll
    for (int g = 0; g < 4; ++g) {
      bv[g] = load16bf(
          (const __bf16*)(lds_b + cur * 4096 + g * 1024 + lane * 32));
    }
    wait_ds0();         // reads of stage cur drained before it can be re-DMAed
    v16bf a_next;
    if (kt + 1 < KT_G) {
      issueB(kt + 1, 1 - cur);   // overlap next DMA with current WMMAs
      a_next = loadA(kt + 1);
    }
#pragma unroll
    for (int g = 0; g < 4; ++g) {
      acc[g] = __builtin_amdgcn_wmma_f32_16x16x32_bf16(
          false, a_cur, false, bv[g], (short)0, acc[g], false, false);
    }
    a_cur = a_next;
  }

  // --- o-gate += xd_t @ Wd (2 k-tiles, direct global) ---
  const __bf16* xdrow = xd_bf + ((size_t)m_row * T_ + t) * E_ + a_koff;
#pragma unroll
  for (int kt = 0; kt < 2; ++kt) {
    v16bf a = load16bf(xdrow + kt * 32);
    const __bf16* bp = Wd_sw + ((size_t)(nt * 2 + kt)) * 512 + lane * 16;
    v16bf bv = load16bf(bp);
    acc[3] = __builtin_amdgcn_wmma_f32_16x16x32_bf16(
        false, a, false, bv, (short)0, acc[3], false, false);
  }

  // --- elementwise gate math; C/D layout: VGPR r -> M = r + (lane>=16)*8,
  //     N = lane&15 ---
  const int n  = nt * 16 + (lane & 15);
  const int m0 = m_base + (lane >> 4) * 8;
  const float bi = bias[0 * H_ + n];
  const float bf = bias[1 * H_ + n];
  const float bg_ = bias[2 * H_ + n];
  const float bo = bias[3 * H_ + n];

#pragma unroll
  for (int r = 0; r < 8; ++r) {
    const int m = m0 + r;
    float gi = sigm(acc[0][r] + bi);
    float gf = sigm(acc[1][r] + bf);
    float gg = tanh_fast(acc[2][r] + bg_);
    float go = sigm(acc[3][r] + bo);
    float c  = gf * c_buf[(size_t)m * H_ + n] + gi * gg;
    float h  = go * tanh_fast(c);
    c_buf[(size_t)m * H_ + n] = c;
    h_out[(size_t)m * H_ + n] = (__bf16)h;
    out_h[((size_t)m * T_ + t) * H_ + n] = h;
  }
}

// Copy h_T (= hidden_seq[:,T-1,:]) and c_T into the tuple tail of d_out.
__global__ void finalize_tail(const float* __restrict__ out_h,
                              const float* __restrict__ c_buf,
                              float* __restrict__ tail) {
  int i = blockIdx.x * blockDim.x + threadIdx.x;
  if (i >= B_ * H_) return;
  int m = i / H_, nn = i % H_;
  tail[i]            = out_h[((size_t)m * T_ + (T_ - 1)) * H_ + nn];
  tail[B_ * H_ + i]  = c_buf[i];
}

// --------------------------- host driver -----------------------------------

extern "C" void kernel_launch(void* const* d_in, const int* in_sizes, int n_in,
                              void* d_out, int out_size, void* d_ws,
                              size_t ws_size, hipStream_t stream) {
  (void)in_sizes; (void)n_in; (void)out_size; (void)ws_size;

  const float* x  = (const float*)d_in[0];
  const float* W  = (const float*)d_in[1];
  const float* U  = (const float*)d_in[2];
  const float* b  = (const float*)d_in[3];
  const float* Wd = (const float*)d_in[4];
  float* out = (float*)d_out;

  char* ws = (char*)d_ws;
  const size_t offG  = 0;                                       // [U;W] swizzled bf16
  const size_t offWd = offG  + (size_t)NT_G * KT_G * 512 * 2;   // + 2,359,296 B
  const size_t offH0 = offWd + (size_t)32 * 2 * 512 * 2;        // +    65,536 B
  const size_t offH1 = offH0 + (size_t)B_ * H_ * 2;             // h ping
  const size_t offC  = offH1 + (size_t)B_ * H_ * 2;             // h pong
  const size_t offX  = offC  + (size_t)B_ * H_ * 4;             // c (f32)
  const size_t offXd = offX  + (size_t)B_ * T_ * E_ * 2;        // x bf16
  // total ~19.7 MB of workspace

  void* G_sw  = ws + offG;
  void* Wd_sw = ws + offWd;
  void* x_bf  = ws + offX;
  void* xd_bf = ws + offXd;
  float* c_buf = (float*)(ws + offC);

  // Prep: swizzle weights, convert activations, zero h0/h1/c (contiguous).
  {
    int nG = NT_G * KT_G * 512;
    swizzle_G<<<(nG + 255) / 256, 256, 0, stream>>>(U, W, G_sw);
    int nWd = 32 * 2 * 512;
    swizzle_Wd<<<(nWd + 255) / 256, 256, 0, stream>>>(Wd, Wd_sw);
    int nX = B_ * T_ * E_;
    conv_x<<<(nX + 255) / 256, 256, 0, stream>>>(x, x_bf, xd_bf);
    int nZ = (int)((2 * (size_t)B_ * H_ * 2 + (size_t)B_ * H_ * 4) / 4);
    zero_u32<<<(nZ + 255) / 256, 256, 0, stream>>>(
        (unsigned int*)(ws + offH0), nZ);
  }

  // Sequential scan: 512 step kernels, h double-buffered across steps.
  for (int t = 0; t < T_; ++t) {
    void* h_in  = (t & 1) ? (void*)(ws + offH1) : (void*)(ws + offH0);
    void* h_out = (t & 1) ? (void*)(ws + offH0) : (void*)(ws + offH1);
    dlstm_step<<<dim3(32), dim3(256), 0, stream>>>(
        G_sw, Wd_sw, x_bf, xd_bf, b, h_in, h_out, c_buf, out, t);
  }

  finalize_tail<<<(B_ * H_ + 255) / 256, 256, 0, stream>>>(
      out, c_buf, out + (size_t)B_ * T_ * H_);
}
